// AspectImp_5540507811970
// MI455X (gfx1250) — matrix operand order repaired and checked by reading, not compiled, not run on hardware
//
#include <hip/hip_runtime.h>

typedef __attribute__((ext_vector_type(16))) __bf16 v16bf;
typedef __attribute__((ext_vector_type(8)))  __bf16 v8bf;
typedef __attribute__((ext_vector_type(4)))  __bf16 v4bf;
typedef __attribute__((ext_vector_type(8)))  float  v8f;

#define NB   128   // batch
#define ASP  64    // aspects
#define HD   256   // H1 == H2
#define LDX  272   // bf16 LDS row stride for 256-wide arrays (16B-aligned rows)
#define LDSM 72    // bf16 LDS row stride for 64-wide arrays (144B rows, 16B-aligned)
#define WST  40    // bf16 row pitch of a staged W k-slice [256][32] (80B, bank-skewed)

// ---- WMMA fragment loaders (CDNA5 v_wmma_f32_16x16x32_bf16; contiguous k order, same for A and B) ----
// A (16x32, MxK): lane L -> M = L%16; elem j -> K = 16*(L/16) + j
// => 16 contiguous bf16, two ds_load_b128.
static __device__ __forceinline__ v16bf load_A_frag(const __bf16* base, int stride,
                                                    int row0, int k0, int lane) {
  const int hlf = lane >> 4;
  const int m   = row0 + (lane & 15);
  const __bf16* p = base + m * stride + k0 + (hlf << 4);
  const v8bf lo = *(const v8bf*)p;         // K = 16*hlf + 0..7
  const v8bf hi = *(const v8bf*)(p + 8);   // K = 16*hlf + 8..15
  return __builtin_shufflevector(lo, hi, 0, 1, 2, 3, 4, 5, 6, 7,
                                 8, 9, 10, 11, 12, 13, 14, 15);
}

// B (32x16, KxN) from a TRANSPOSED store baseT[N][K]:
// lane L -> N = L%16; elem j -> K = 16*(L/16) + j => 16 contiguous bf16.
static __device__ __forceinline__ v16bf load_B_fragT(const __bf16* baseT, int stride,
                                                     int k0, int col0, int lane) {
  const __bf16* p = baseT + (col0 + (lane & 15)) * stride + k0 + ((lane >> 4) << 4);
  const v8bf lo = *(const v8bf*)p;
  const v8bf hi = *(const v8bf*)(p + 8);
  return __builtin_shufflevector(lo, hi, 0, 1, 2, 3, 4, 5, 6, 7,
                                 8, 9, 10, 11, 12, 13, 14, 15);
}

// Stage one 32-deep k-slice of Wu and Wi into LDS as bf16, layout [k2][h_local]
static __device__ __forceinline__ void stage_W(__bf16* Wst, int buf,
                                               const float* __restrict__ Wu,
                                               const float* __restrict__ Wi,
                                               int ks, int tid) {
  __bf16* du = Wst + (buf * 2 + 0) * (HD * WST);
  __bf16* di = Wst + (buf * 2 + 1) * (HD * WST);
#pragma unroll
  for (int q = 0; q < 8; ++q) {
    const int e  = tid * 32 + q * 4;         // 8192 elems per matrix per slice
    const int k2 = e >> 5, hl = e & 31;
    const float4 fu = *(const float4*)(Wu + k2 * HD + ks * 32 + hl);
    const float4 fi = *(const float4*)(Wi + k2 * HD + ks * 32 + hl);
    const v4bf bu = {(__bf16)fu.x, (__bf16)fu.y, (__bf16)fu.z, (__bf16)fu.w};
    const v4bf bi = {(__bf16)fi.x, (__bf16)fi.y, (__bf16)fi.z, (__bf16)fi.w};
    *(v4bf*)&du[k2 * WST + hl] = bu;
    *(v4bf*)&di[k2 * WST + hl] = bi;
  }
}

__global__ __launch_bounds__(256)
void aspect_importance_kernel(const float* __restrict__ Uasp,
                              const float* __restrict__ Iasp,
                              const float* __restrict__ Wu,
                              const float* __restrict__ Wi,
                              const float* __restrict__ whu,
                              const float* __restrict__ whi,
                              float* __restrict__ out) {
  extern __shared__ char smem[];
  __bf16* Ubf   = (__bf16*)smem;            // [64][272]
  __bf16* Ibf   = Ubf  + ASP * LDX;         // [64][272]
  __bf16* PuT   = Ibf  + ASP * LDX;         // [256][72]  (P_u transposed: [k2][u])
  __bf16* PiT   = PuT  + HD * LDSM;         // [256][72]
  __bf16* simb  = PiT  + HD * LDSM;         // [64][72]
  __bf16* simTb = simb + ASP * LDSM;        // [64][72]
  __bf16* Wst   = simTb + ASP * LDSM;       // [2 buf][2 mat][256][40]
  float* whu_s  = (float*)(Wst + 4 * HD * WST);
  float* whi_s  = whu_s + HD;
  float* scores = whi_s + HD;               // [128] user|item
  float* red    = scores + 2 * ASP;         // [4]

  const int b    = blockIdx.x;
  const int tid  = threadIdx.x;
  const int lane = tid & 31;
  const int wave = tid >> 5;

  const float* Ug = Uasp + (size_t)b * ASP * HD;
  const float* Ig = Iasp + (size_t)b * ASP * HD;

  // Warm the shared weight matrices into L2 (global_prefetch_b8)
  for (int off = tid * 64; off < HD * HD; off += 256 * 64) {
    __builtin_prefetch(Wu + off, 0, 1);
    __builtin_prefetch(Wi + off, 0, 1);
  }

  // Stage aspect representations into LDS as bf16 (float4 in, packed v4bf out)
  for (int idx = tid * 4; idx < ASP * HD; idx += 256 * 4) {
    const float4 fu = *(const float4*)(Ug + idx);
    const float4 fi = *(const float4*)(Ig + idx);
    const int r = idx >> 8, c = idx & (HD - 1);
    const v4bf bu = {(__bf16)fu.x, (__bf16)fu.y, (__bf16)fu.z, (__bf16)fu.w};
    const v4bf bi = {(__bf16)fi.x, (__bf16)fi.y, (__bf16)fi.z, (__bf16)fi.w};
    *(v4bf*)&Ubf[r * LDX + c] = bu;
    *(v4bf*)&Ibf[r * LDX + c] = bi;
  }
  whu_s[tid] = whu[tid];
  whi_s[tid] = whi[tid];
  if (tid < 2 * ASP) scores[tid] = 0.0f;
  __syncthreads();

  // ---- Phase 1: L1 pairwise similarity (VALU; b128 LDS traffic) ----
  {
    const int u  = tid >> 2;          // 1 user row x 16 item rows per thread
    const int i0 = (tid & 3) << 4;
    float s[16];
#pragma unroll
    for (int ii = 0; ii < 16; ++ii) s[ii] = 0.0f;
    for (int h0 = 0; h0 < HD; h0 += 8) {
      const v8bf u8 = *(const v8bf*)&Ubf[u * LDX + h0];
#pragma unroll
      for (int ii = 0; ii < 16; ++ii) {
        const v8bf i8 = *(const v8bf*)&Ibf[(i0 + ii) * LDX + h0];
#pragma unroll
        for (int e = 0; e < 8; ++e)
          s[ii] += fabsf((float)u8[e] - (float)i8[e]);
      }
    }
    v8bf sv0, sv1;
#pragma unroll
    for (int ii = 0; ii < 16; ++ii) {
      const float v  = 1.0f / (1.0f + s[ii]);
      const __bf16 vb = (__bf16)v;
      if (ii < 8) sv0[ii] = vb; else sv1[ii - 8] = vb;
      simTb[(i0 + ii) * LDSM + u] = vb;
    }
    *(v8bf*)&simb[u * LDSM + i0]     = sv0;
    *(v8bf*)&simb[u * LDSM + i0 + 8] = sv1;
  }

  // ---- Phase 2: projections P = X @ W^T via WMMA ----
  // W k-slices staged once per WG through LDS (bf16), double buffered.
  const int isItem = wave >> 2;                 // waves 0-3: user, 4-7: item
  const int rtile  = (wave & 3) * 16;           // 16-row tile owned by this wave
  const __bf16* Xbf = isItem ? Ibf : Ubf;

  const v8f vzero = {0.f, 0.f, 0.f, 0.f, 0.f, 0.f, 0.f, 0.f};
  v8f acc[16];
#pragma unroll
  for (int t = 0; t < 16; ++t) acc[t] = vzero;

  stage_W(Wst, 0, Wu, Wi, 0, tid);
  __syncthreads();

  for (int ks = 0; ks < 8; ++ks) {              // K = 256 in steps of 32
    const __bf16* WS = Wst + ((ks & 1) * 2 + isItem) * (HD * WST);
    const v16bf a = load_A_frag(Xbf, LDX, rtile, ks * 32, lane);
    if (ks + 1 < 8) stage_W(Wst, (ks + 1) & 1, Wu, Wi, ks + 1, tid);
    // Depth-2 software pipeline over the 16 column tiles. sched_barrier(0)
    // pins program order L(t+2) || W(t): the scheduler cannot sink the loads
    // back onto their consumers, forcing 3 live fragment octets and partial
    // s_wait_dscnt instead of full serialization.
    v16bf b0 = load_B_fragT(WS, WST, 0, 0,  lane);
    v16bf b1 = load_B_fragT(WS, WST, 0, 16, lane);
#pragma unroll
    for (int t = 0; t < 16; ++t) {
      v16bf bn = b0;
      if (t + 2 < 16) bn = load_B_fragT(WS, WST, 0, (t + 2) * 16, lane);
      __builtin_amdgcn_sched_barrier(0);
      acc[t] = __builtin_amdgcn_wmma_f32_16x16x32_bf16(
          false, a, false, b0, (short)0, acc[t], false, false);
      __builtin_amdgcn_sched_barrier(0);
      b0 = b1;
      b1 = bn;
    }
    __syncthreads();   // next slice staged; current slice free for overwrite
  }

  // Spill P (transposed, bf16) to LDS: one packed 16B store per tile
  const int hlf = lane >> 4, n = lane & 15;
#pragma unroll
  for (int t = 0; t < 16; ++t) {
    v8bf pk;
#pragma unroll
    for (int v = 0; v < 8; ++v) pk[v] = (__bf16)acc[t][v];
    *(v8bf*)&((isItem ? PiT : PuT)[(t * 16 + n) * LDSM + rtile + (hlf << 3)]) = pk;
  }
  __syncthreads();

  // ---- Phase 3: H = relu(P + sim @ P_other), score = w . H  (WMMA chains on acc) ----
  const __bf16* Afrag = isItem ? simTb : simb;
  const __bf16* BsrcT = isItem ? PuT : PiT;
  const float*  wv    = isItem ? whi_s : whu_s;

  const v16bf a0 = load_A_frag(Afrag, LDSM, rtile, 0,  lane);
  const v16bf a1 = load_A_frag(Afrag, LDSM, rtile, 32, lane);

  float partial[8];
#pragma unroll
  for (int v = 0; v < 8; ++v) partial[v] = 0.0f;

  // Depth-1 pipeline: next tile's (b0,b1) loads issue under the two chained WMMAs.
  v16bf pb0 = load_B_fragT(BsrcT, LDSM, 0,  0, lane);
  v16bf pb1 = load_B_fragT(BsrcT, LDSM, 32, 0, lane);
#pragma unroll
  for (int t = 0; t < 16; ++t) {
    v16bf nb0 = pb0, nb1 = pb1;
    if (t + 1 < 16) {
      nb0 = load_B_fragT(BsrcT, LDSM, 0,  (t + 1) * 16, lane);
      nb1 = load_B_fragT(BsrcT, LDSM, 32, (t + 1) * 16, lane);
    }
    __builtin_amdgcn_sched_barrier(0);
    v8f c = __builtin_amdgcn_wmma_f32_16x16x32_bf16(
        false, a0, false, pb0, (short)0, acc[t], false, false);  // C = P tile (f32)
    c = __builtin_amdgcn_wmma_f32_16x16x32_bf16(
        false, a1, false, pb1, (short)0, c, false, false);
    const float w = wv[t * 16 + n];
#pragma unroll
    for (int v = 0; v < 8; ++v) partial[v] += fmaxf(c[v], 0.0f) * w;
    __builtin_amdgcn_sched_barrier(0);
    pb0 = nb0;
    pb1 = nb1;
  }
  {
    float* sc = scores + isItem * ASP;
#pragma unroll
    for (int v = 0; v < 8; ++v)
      atomicAdd(&sc[rtile + v + 8 * hlf], partial[v]);   // ds_add_f32
  }
  __syncthreads();

  // ---- Phase 4: softmax over 64 aspects (user & item) ----
  if (tid < 2) {
    const float* s = scores + tid * ASP;
    float m = -3.0e38f;
    for (int j = 0; j < ASP; ++j) m = fmaxf(m, s[j]);
    float sum = 0.0f;
    for (int j = 0; j < ASP; ++j) sum += __expf(s[j] - m);
    red[tid * 2 + 0] = m;
    red[tid * 2 + 1] = sum;
  }
  __syncthreads();
  if (tid < 2 * ASP) {
    const int which = tid >> 6;              // 0 = user, 1 = item
    const int j     = tid & (ASP - 1);
    const float m = red[which * 2], sum = red[which * 2 + 1];
    out[(size_t)which * NB * ASP + (size_t)b * ASP + j] =
        __expf(scores[which * ASP + j] - m) / sum;
  }
}

extern "C" void kernel_launch(void* const* d_in, const int* in_sizes, int n_in,
                              void* d_out, int out_size, void* d_ws, size_t ws_size,
                              hipStream_t stream) {
  (void)in_sizes; (void)n_in; (void)out_size; (void)d_ws; (void)ws_size;
  const float* U   = (const float*)d_in[0];
  const float* I   = (const float*)d_in[1];
  const float* Wu  = (const float*)d_in[2];
  const float* Wi  = (const float*)d_in[3];
  const float* whu = (const float*)d_in[4];
  const float* whi = (const float*)d_in[5];
  float* out = (float*)d_out;

  const size_t smem =
      (size_t)(2 * ASP * LDX + 2 * HD * LDSM + 2 * ASP * LDSM + 4 * HD * WST) * 2
      + (size_t)(2 * HD + 2 * ASP + 4) * sizeof(float);
  (void)hipFuncSetAttribute((const void*)aspect_importance_kernel,
                            hipFuncAttributeMaxDynamicSharedMemorySize, (int)smem);
  aspect_importance_kernel<<<dim3(NB), dim3(256), smem, stream>>>(
      U, I, Wu, Wi, whu, whi, out);
}
